// PairformerLayer_66589172957489
// MI455X (gfx1250) — compile-verified, matrix operand order checked by reading, not run on hardware
//
#include <hip/hip_runtime.h>
#include <math.h>

typedef __bf16 bf16;
typedef __attribute__((ext_vector_type(16))) __bf16 v16bf;
typedef __attribute__((ext_vector_type(8)))  __bf16 v8bf;
typedef __attribute__((ext_vector_type(8)))  float  v8f;

enum {
  M_F32 = 0,          // Cf = acc*scale
  M_BF16,             // Cb = acc
  M_SIG_F32,          // Cf = sigmoid(acc)
  M_DUAL_SIG_BF16,    // Cb = sigmoid(acc1)*acc2
  M_DUAL_SILU_BF16,   // Cb = silu(acc1)*acc2
  M_GATE_ADD,         // Cf[idx] += gate[idx]*acc
  M_ADD,              // Cf[idx] += acc
  M_ADD_T,            // Cf[transposed row] += acc   (row in [0,65536), N=256)
  M_LOGITS,           // Cf = acc*scale + bias16[(row*256+col)*16 + inner]
  M_PAD_BF16          // Cb[row*512 + (col/24)*32 + col%24] = acc + bias[col]
};

// ---------------------------------------------------------------------------
// WMMA fragment load: A is [M,K] row-major, B supplied as Wt [Nc,K] row-major.
// 16-bit A 16x32 layout: lanes 0-15 row=lane, K {0..7,16..23}; lanes 16-31
// row=lane-16, K {8..15,24..31}. B mirrors with columns as "rows" of Wt.
// ---------------------------------------------------------------------------
__device__ __forceinline__ v16bf load_frag(const bf16* __restrict__ base,
                                           int ld, int row, int k0) {
  int lane = threadIdx.x & 31;
  int r    = row + (lane & 15);
  int koff = k0 + ((lane >> 4) << 3);          // 0 or 8
  const bf16* p = base + (size_t)r * (size_t)ld + koff;
  union { v16bf v; v8bf h[2]; } u;
  u.h[0] = *(const v8bf*)(p);
  u.h[1] = *(const v8bf*)(p + 16);
  return u.v;
}

__device__ __forceinline__ v8f wmma_bf(v16bf a, v16bf b, v8f c) {
  return __builtin_amdgcn_wmma_f32_16x16x32_bf16(false, a, false, b,
                                                 (short)0, c, false, false);
}

// ---- branch-free, software-pipelined K-loops ------------------------------
// 2 m-tiles x 2 n-tiles, single weight matrix
__device__ __forceinline__ void mma_n2(const bf16* __restrict__ A,
                                       const bf16* __restrict__ B,
                                       int lda, int ldb, int m0, int n0, int K,
                                       v8f acc[4]) {
  v16bf a0 = load_frag(A, lda, m0, 0);
  v16bf a1 = load_frag(A, lda, m0 + 16, 0);
  v16bf b0 = load_frag(B, ldb, n0, 0);
  v16bf b1 = load_frag(B, ldb, n0 + 16, 0);
  for (int k0 = 32; k0 < K; k0 += 32) {
    v16bf na0 = load_frag(A, lda, m0, k0);
    v16bf na1 = load_frag(A, lda, m0 + 16, k0);
    v16bf nb0 = load_frag(B, ldb, n0, k0);
    v16bf nb1 = load_frag(B, ldb, n0 + 16, k0);
    acc[0] = wmma_bf(a0, b0, acc[0]);
    acc[1] = wmma_bf(a1, b0, acc[1]);
    acc[2] = wmma_bf(a0, b1, acc[2]);
    acc[3] = wmma_bf(a1, b1, acc[3]);
    a0 = na0; a1 = na1; b0 = nb0; b1 = nb1;
  }
  acc[0] = wmma_bf(a0, b0, acc[0]);
  acc[1] = wmma_bf(a1, b0, acc[1]);
  acc[2] = wmma_bf(a0, b1, acc[2]);
  acc[3] = wmma_bf(a1, b1, acc[3]);
}

// 2 m-tiles x 1 n-tile (narrow Nc)
__device__ __forceinline__ void mma_n1(const bf16* __restrict__ A,
                                       const bf16* __restrict__ B,
                                       int lda, int ldb, int m0, int n0, int K,
                                       v8f acc[4]) {
  v16bf a0 = load_frag(A, lda, m0, 0);
  v16bf a1 = load_frag(A, lda, m0 + 16, 0);
  v16bf b0 = load_frag(B, ldb, n0, 0);
  for (int k0 = 32; k0 < K; k0 += 32) {
    v16bf na0 = load_frag(A, lda, m0, k0);
    v16bf na1 = load_frag(A, lda, m0 + 16, k0);
    v16bf nb0 = load_frag(B, ldb, n0, k0);
    acc[0] = wmma_bf(a0, b0, acc[0]);
    acc[1] = wmma_bf(a1, b0, acc[1]);
    a0 = na0; a1 = na1; b0 = nb0;
  }
  acc[0] = wmma_bf(a0, b0, acc[0]);
  acc[1] = wmma_bf(a1, b0, acc[1]);
}

// 2 m-tiles x 1 n-tile, two weight matrices (dual/gated GEMMs)
// acc[0..1] <- A.B ; acc[2..3] <- A.B2
__device__ __forceinline__ void mma_dual(const bf16* __restrict__ A,
                                         const bf16* __restrict__ B,
                                         const bf16* __restrict__ B2,
                                         int lda, int ldb, int m0, int n0, int K,
                                         v8f acc[4]) {
  v16bf a0 = load_frag(A, lda, m0, 0);
  v16bf a1 = load_frag(A, lda, m0 + 16, 0);
  v16bf b0 = load_frag(B, ldb, n0, 0);
  v16bf c0 = load_frag(B2, ldb, n0, 0);
  for (int k0 = 32; k0 < K; k0 += 32) {
    v16bf na0 = load_frag(A, lda, m0, k0);
    v16bf na1 = load_frag(A, lda, m0 + 16, k0);
    v16bf nb0 = load_frag(B, ldb, n0, k0);
    v16bf nc0 = load_frag(B2, ldb, n0, k0);
    acc[0] = wmma_bf(a0, b0, acc[0]);
    acc[1] = wmma_bf(a1, b0, acc[1]);
    acc[2] = wmma_bf(a0, c0, acc[2]);
    acc[3] = wmma_bf(a1, c0, acc[3]);
    a0 = na0; a1 = na1; b0 = nb0; c0 = nc0;
  }
  acc[0] = wmma_bf(a0, b0, acc[0]);
  acc[1] = wmma_bf(a1, b0, acc[1]);
  acc[2] = wmma_bf(a0, c0, acc[2]);
  acc[3] = wmma_bf(a1, c0, acc[3]);
}

// Epilogue for one 16x16 tile. C/D layout: VGPR g -> row m0+g (+8 upper half),
// col n0+(lane&15).
__device__ __forceinline__ void store_tile(
    int mode, v8f acc, v8f acc2, int m0, int n0,
    int M, int Nc, int ldc, int cstride, long long cOff,
    float* Cf, bf16* Cb, const float* __restrict__ bias,
    const float* __restrict__ gate, const float* __restrict__ bias16,
    int inner, float scale) {
  int lane = threadIdx.x & 31;
#pragma unroll
  for (int g = 0; g < 8; ++g) {
    int row = m0 + g + ((lane >> 4) << 3);
    int col = n0 + (lane & 15);
    if (row >= M || col >= Nc) continue;
    float r = acc[g];
    long long idx = cOff + (long long)row * ldc + (long long)col * cstride;
    switch (mode) {
      case M_F32:      Cf[idx] = r * scale; break;
      case M_BF16:     Cb[idx] = (bf16)r;   break;
      case M_SIG_F32:  Cf[idx] = 1.f / (1.f + __expf(-r)); break;
      case M_DUAL_SIG_BF16:
        Cb[idx] = (bf16)((1.f / (1.f + __expf(-r))) * acc2[g]); break;
      case M_DUAL_SILU_BF16:
        Cb[idx] = (bf16)((r / (1.f + __expf(-r))) * acc2[g]); break;
      case M_GATE_ADD: Cf[idx] += gate[idx] * r; break;
      case M_ADD:      Cf[idx] += r; break;
      case M_ADD_T: {
        long long tr = (long long)((row & 255) * 256 + (row >> 8)) * ldc
                     + (long long)col * cstride;
        Cf[tr] += r;
      } break;
      case M_LOGITS:
        Cf[idx] = r * scale
                + bias16[((long long)row * 256 + col) * 16 + inner];
        break;
      case M_PAD_BF16: {
        int pc = (col / 24) * 32 + (col % 24);
        float v = r + (bias ? bias[col] : 0.f);
        Cb[(long long)row * 512 + pc] = (bf16)v;
      } break;
    }
  }
}

// ---------------------------------------------------------------------------
// Generic batched WMMA GEMM.  Block = 256 threads = 8 waves (2 m x 4 n).
// Wave tile: 32x32 (normal) or 32x16 with two weight matrices (dual).
// Block tile: 64x128 (normal) / 64x64 (dual).
// batch bz: outer = bz/innerCnt, inner = bz%innerCnt; element offsets
// outer*{a,b,c}Out + inner*{a,b,c}In applied to A, B, C.
// C element index = cOff + row*ldc + col*cstride  (modes override as noted).
// ---------------------------------------------------------------------------
__global__ void wmma_gemm(const bf16* __restrict__ A,
                          const bf16* __restrict__ B,
                          const bf16* __restrict__ B2,
                          float* Cf, bf16* Cb,
                          const float* __restrict__ bias,
                          const float* __restrict__ gate,
                          const float* __restrict__ bias16,
                          int M, int Nc, int NcPad, int K,
                          int lda, int ldb, int ldc, int cstride,
                          int innerCnt,
                          long long aOut, long long aIn,
                          long long bOut, long long bIn,
                          long long cOut, long long cIn,
                          float scale, int mode) {
  int wave = (int)(threadIdx.x >> 5);
  bool dual = (mode == M_DUAL_SIG_BF16) || (mode == M_DUAL_SILU_BF16);
  int nspan = dual ? 16 : 32;
  int m0 = blockIdx.x * 64 + (wave >> 2) * 32;
  int n0 = blockIdx.y * (nspan * 4) + (wave & 3) * nspan;
  if (m0 >= M || n0 >= NcPad) return;          // wave-uniform exit

  int bz    = (int)blockIdx.z;
  int outer = bz / innerCnt;
  int inner = bz - outer * innerCnt;
  A += outer * aOut + inner * aIn;
  B += outer * bOut + inner * bIn;
  const bf16* B2b = B2;
  long long cOff = outer * cOut + inner * cIn;

  v8f acc[4];
#pragma unroll
  for (int t = 0; t < 4; ++t) acc[t] = v8f{0.f,0.f,0.f,0.f,0.f,0.f,0.f,0.f};

  bool n1 = !dual && (n0 + 32 <= NcPad);
  if (dual)      mma_dual(A, B, B2b, lda, ldb, m0, n0, K, acc);
  else if (n1)   mma_n2(A, B, lda, ldb, m0, n0, K, acc);
  else           mma_n1(A, B, lda, ldb, m0, n0, K, acc);

  if (dual) {
    store_tile(mode, acc[0], acc[2], m0, n0, M, Nc, ldc, cstride, cOff,
               Cf, Cb, bias, gate, bias16, inner, scale);
    store_tile(mode, acc[1], acc[3], m0 + 16, n0, M, Nc, ldc, cstride, cOff,
               Cf, Cb, bias, gate, bias16, inner, scale);
  } else {
    store_tile(mode, acc[0], acc[0], m0, n0, M, Nc, ldc, cstride, cOff,
               Cf, Cb, bias, gate, bias16, inner, scale);
    store_tile(mode, acc[1], acc[1], m0 + 16, n0, M, Nc, ldc, cstride, cOff,
               Cf, Cb, bias, gate, bias16, inner, scale);
    if (n1) {
      store_tile(mode, acc[2], acc[2], m0, n0 + 16, M, Nc, ldc, cstride, cOff,
                 Cf, Cb, bias, gate, bias16, inner, scale);
      store_tile(mode, acc[3], acc[3], m0 + 16, n0 + 16, M, Nc, ldc, cstride,
                 cOff, Cf, Cb, bias, gate, bias16, inner, scale);
    }
  }
}

// ---------------------------------------------------------------------------
// LayerNorm over last dim D (block per row, blockDim == D <= 512).
// transposeN != 0 -> read source row (j*transposeN + i) for logical row (i,j).
// ---------------------------------------------------------------------------
__global__ void ln_kernel(const float* __restrict__ src,
                          const float* __restrict__ gamma,
                          const float* __restrict__ beta,
                          bf16* __restrict__ dst, int D, int transposeN) {
  extern __shared__ float sm[];
  int r = (int)blockIdx.x;
  int c = (int)threadIdx.x;
  long long srow;
  if (transposeN) {
    int i = r / transposeN, j = r - i * transposeN;
    srow = (long long)j * transposeN + i;
  } else {
    srow = r;
  }
  float x = src[srow * D + c];
  sm[c] = x; __syncthreads();
  for (int s = 256; s > 0; s >>= 1) {
    if (c < s && c + s < D) sm[c] += sm[c + s];
    __syncthreads();
  }
  float mean = sm[0] / (float)D; __syncthreads();
  float d = x - mean;
  sm[c] = d * d; __syncthreads();
  for (int s = 256; s > 0; s >>= 1) {
    if (c < s && c + s < D) sm[c] += sm[c + s];
    __syncthreads();
  }
  float var = sm[0] / (float)D;
  float y = d * rsqrtf(var + 1e-5f) * gamma[c] + beta[c];
  dst[(long long)r * D + c] = (bf16)y;
}

// Softmax over rows of length 256, f32 in -> bf16 out.
__global__ void softmax_kernel(const float* __restrict__ logits,
                               bf16* __restrict__ probs) {
  __shared__ float sm[256];
  long long r = blockIdx.x;
  int c = (int)threadIdx.x;
  float x = logits[r * 256 + c];
  sm[c] = x; __syncthreads();
  for (int s = 128; s > 0; s >>= 1) {
    if (c < s) sm[c] = fmaxf(sm[c], sm[c + s]);
    __syncthreads();
  }
  float mx = sm[0]; __syncthreads();
  float e = __expf(x - mx);
  sm[c] = e; __syncthreads();
  for (int s = 128; s > 0; s >>= 1) {
    if (c < s) sm[c] += sm[c + s];
    __syncthreads();
  }
  probs[r * 256 + c] = (bf16)(e / sm[0]);
}

// Weight convert: src f32 [K,Nc] row-major -> dst bf16 [NcPad,K] (Wt), pad=0.
__global__ void wconv(const float* __restrict__ src, bf16* __restrict__ dst,
                      int K, int Nc, int NcPad) {
  long long idx = (long long)blockIdx.x * blockDim.x + threadIdx.x;
  long long tot = (long long)NcPad * K;
  if (idx >= tot) return;
  int k = (int)(idx % K);
  int n = (int)(idx / K);
  dst[idx] = (n < Nc) ? (bf16)src[(long long)k * Nc + n] : (bf16)0.f;
}

// Triangle einsum staging: ab bf16 [65536,256] -> at/bt bf16 [128][256][256].
// outgoing: at[d][r][c] = ab[(r*256+c)*256 + d],     bt ... col 128+d
// incoming: source row = c*256 + r.
__global__ void stage_ab(const bf16* __restrict__ ab,
                         bf16* __restrict__ at, bf16* __restrict__ bt,
                         int outgoing) {
  long long idx = (long long)blockIdx.x * blockDim.x + threadIdx.x;
  if (idx >= (long long)128 * 65536) return;
  int c = (int)(idx & 255);
  int r = (int)((idx >> 8) & 255);
  int d = (int)(idx >> 16);
  long long srow = outgoing ? (long long)r * 256 + c : (long long)c * 256 + r;
  at[idx] = ab[srow * 256 + d];
  bt[idx] = ab[srow * 256 + 128 + d];
}

// Triangle V transpose: vt[((i*4+h)*32+d)*256+k] = v[(i*256+k)*128 + h*32+d]
__global__ void vt_tri(const bf16* __restrict__ v, bf16* __restrict__ vt) {
  long long idx = (long long)blockIdx.x * blockDim.x + threadIdx.x;
  if (idx >= (long long)65536 * 128) return;
  int k = (int)(idx & 255);
  int d = (int)((idx >> 8) & 31);
  int h = (int)((idx >> 13) & 3);
  int i = (int)(idx >> 15);
  vt[idx] = v[((long long)i * 256 + k) * 128 + h * 32 + d];
}

// s-attention V transpose (head dim 24 padded to 32):
// vt[(h*32+d)*256+k] = d<24 ? v[k*384 + h*24 + d] : 0
__global__ void vt_apb(const bf16* __restrict__ v, bf16* __restrict__ vt) {
  int idx = (int)(blockIdx.x * blockDim.x + threadIdx.x);
  if (idx >= 16 * 32 * 256) return;
  int k = idx & 255;
  int d = (idx >> 8) & 31;
  int h = idx >> 13;
  vt[idx] = (d < 24) ? v[k * 384 + h * 24 + d] : (bf16)0.f;
}

__global__ void gate_mul(const float* __restrict__ t,
                         const float* __restrict__ g,
                         bf16* __restrict__ out, long long n) {
  long long idx = (long long)blockIdx.x * blockDim.x + threadIdx.x;
  if (idx < n) out[idx] = (bf16)(t[idx] * g[idx]);
}

__global__ void copy_f32(float* __restrict__ dst, const float* __restrict__ src,
                         long long n) {
  long long idx = (long long)blockIdx.x * blockDim.x + threadIdx.x;
  if (idx < n) dst[idx] = src[idx];
}

__global__ void zero_bf16(bf16* __restrict__ p, long long n) {
  long long idx = (long long)blockIdx.x * blockDim.x + threadIdx.x;
  if (idx < n) p[idx] = (bf16)0.f;
}

// ---------------------------------------------------------------------------
extern "C" void kernel_launch(void* const* d_in, const int* in_sizes, int n_in,
                              void* d_out, int out_size, void* d_ws, size_t ws_size,
                              hipStream_t stream) {
  (void)in_sizes; (void)n_in; (void)out_size; (void)ws_size;
  const float* P[64];
  for (int i = 0; i < 64 && i < n_in; ++i) P[i] = (const float*)d_in[i];

  const float* s_in = P[0];
  const float* z_in = P[1];
  float* s_out = (float*)d_out;                 // [256,384]
  float* z_out = s_out + 98304;                 // [65536,128]

  char* ws = (char*)d_ws;
  const size_t MB = (size_t)1 << 20;
  const size_t OFF_W    = 0;                    // bf16 weights (~6.2 MB)
  const size_t OFF_XN   = 8 * MB;               // bf16 [65536,128]
  const size_t OFF_WIDE = 24 * MB;              // bf16 up to [65536,512]
  const size_t OFF_AT   = 88 * MB;              // bf16 [128,256,256]
  const size_t OFF_BT   = 104 * MB;             // bf16 [128,256,256]
  const size_t OFF_VT   = 120 * MB;             // bf16 [65536,128]
  const size_t OFF_T    = 136 * MB;             // f32  [65536,128]
  const size_t OFF_G    = 168 * MB;             // f32  [65536,128]
  const size_t OFF_BN   = 200 * MB;             // bf16 [65536,128]
  const size_t OFF_B16  = 216 * MB;             // f32  [65536,16]
  const size_t OFF_LOG  = 220 * MB;             // f32  [16*4*256,256]
  const size_t OFF_PRB  = 237 * MB;             // bf16 [16*4*256,256]
  const size_t OFF_MISC = 246 * MB;

  bf16*  XN  = (bf16*)(ws + OFF_XN);
  bf16*  WD  = (bf16*)(ws + OFF_WIDE);
  bf16*  AT  = (bf16*)(ws + OFF_AT);
  bf16*  BT  = (bf16*)(ws + OFF_BT);
  bf16*  VT  = (bf16*)(ws + OFF_VT);
  float* Tb  = (float*)(ws + OFF_T);
  float* Gb  = (float*)(ws + OFF_G);
  bf16*  BN  = (bf16*)(ws + OFF_BN);
  float* B16 = (float*)(ws + OFF_B16);
  float* LOG = (float*)(ws + OFF_LOG);
  bf16*  PRB = (bf16*)(ws + OFF_PRB);

  size_t woff = OFF_W;
  auto conv = [&](const float* src, int K, int Nc, int NcPad) -> bf16* {
    bf16* dst = (bf16*)(ws + woff);
    woff += (size_t)NcPad * K * sizeof(bf16);
    woff = (woff + 255) & ~(size_t)255;
    long long n = (long long)NcPad * K;
    wconv<<<dim3((unsigned)((n + 255) / 256)), dim3(256), 0, stream>>>(src, dst, K, Nc, NcPad);
    return dst;
  };

  auto gemm = [&](const bf16* A, const bf16* B, const bf16* B2,
                  float* Cf, bf16* Cb,
                  const float* bias, const float* gate, const float* b16,
                  int M, int Nc, int NcPad, int K,
                  int lda, int ldb, int ldc, int cstride,
                  int nbatch, int innerCnt,
                  long long aOut, long long aIn, long long bOut, long long bIn,
                  long long cOut, long long cIn, float scale, int mode) {
    bool dual = (mode == M_DUAL_SIG_BF16) || (mode == M_DUAL_SILU_BF16);
    int bn = dual ? 64 : 128;
    dim3 grid((unsigned)((M + 63) / 64), (unsigned)((NcPad + bn - 1) / bn),
              (unsigned)nbatch);
    wmma_gemm<<<grid, dim3(256), 0, stream>>>(A, B, B2, Cf, Cb, bias, gate, b16,
                                              M, Nc, NcPad, K, lda, ldb, ldc,
                                              cstride, innerCnt, aOut, aIn,
                                              bOut, bIn, cOut, cIn, scale, mode);
  };

  // ---- convert all weights to bf16 transposed [NcPad, K] ------------------
  bf16 *tmo_wp = conv(P[4], 128, 256, 256), *tmo_wgp = conv(P[5], 128, 256, 256);
  bf16 *tmo_wo = conv(P[8], 128, 128, 128), *tmo_wgo = conv(P[9], 128, 128, 128);
  bf16 *tmi_wp = conv(P[12], 128, 256, 256), *tmi_wgp = conv(P[13], 128, 256, 256);
  bf16 *tmi_wo = conv(P[16], 128, 128, 128), *tmi_wgo = conv(P[17], 128, 128, 128);
  bf16 *tas_wb = conv(P[20], 128, 4, 16);
  bf16 *tas_wq = conv(P[21], 128, 128, 128), *tas_wk = conv(P[22], 128, 128, 128);
  bf16 *tas_wv = conv(P[23], 128, 128, 128), *tas_wg = conv(P[24], 128, 128, 128);
  bf16 *tas_wo = conv(P[25], 128, 128, 128);
  bf16 *tae_wb = conv(P[28], 128, 4, 16);
  bf16 *tae_wq = conv(P[29], 128, 128, 128), *tae_wk = conv(P[30], 128, 128, 128);
  bf16 *tae_wv = conv(P[31], 128, 128, 128), *tae_wg = conv(P[32], 128, 128, 128);
  bf16 *tae_wo = conv(P[33], 128, 128, 128);
  bf16 *tz_w1 = conv(P[36], 128, 512, 512), *tz_w2 = conv(P[37], 128, 512, 512);
  bf16 *tz_w3 = conv(P[38], 512, 128, 128);
  bf16 *at_wq = conv(P[41], 384, 384, 384), *at_wk = conv(P[43], 384, 384, 384);
  bf16 *at_wv = conv(P[44], 384, 384, 384), *at_wg = conv(P[45], 384, 384, 384);
  bf16 *at_wz = conv(P[48], 128, 16, 16),   *at_wo = conv(P[49], 384, 384, 384);
  bf16 *ts_w1 = conv(P[52], 384, 1536, 1536), *ts_w2 = conv(P[53], 384, 1536, 1536);
  bf16 *ts_w3 = conv(P[54], 1536, 384, 384);

  // ---- residual accumulators in d_out -------------------------------------
  copy_f32<<<dim3(98304 / 256), dim3(256), 0, stream>>>(s_out, s_in, 98304);
  copy_f32<<<dim3(8388608 / 256), dim3(256), 0, stream>>>(z_out, z_in, 8388608);

  const long long NN = 65536;

  // ---- triangle multiply (outgoing then incoming) -------------------------
  auto tri_mul = [&](const float* lnw, const float* lnb, bf16* wp, bf16* wgp,
                     const float* lnow, const float* lnob, bf16* wo, bf16* wgo,
                     int outgoing) {
    ln_kernel<<<dim3(65536), dim3(128), 128 * 4, stream>>>(z_out, lnw, lnb, XN, 128, 0);
    // ab = sigmoid(xn@w_gp) * (xn@w_p)  -> bf16 [65536,256]
    gemm(XN, wgp, wp, nullptr, WD, nullptr, nullptr, nullptr,
         65536, 256, 256, 128, 128, 128, 256, 1,
         1, 1, 0, 0, 0, 0, 0, 0, 1.f, M_DUAL_SIG_BF16);
    stage_ab<<<dim3(8388608 / 256), dim3(256), 0, stream>>>(WD, AT, BT, outgoing);
    // t[i,j,d] = sum_k at[d][i][k]*bt[d][j][k]  (128 batched 256x256x256)
    gemm(AT, BT, nullptr, Tb, nullptr, nullptr, nullptr, nullptr,
         256, 256, 256, 256, 256, 256, 256 * 128, 128,
         128, 1, NN, 0, NN, 0, 1, 0, 1.f, M_F32);
    // gate = sigmoid(xn@w_go)
    gemm(XN, wgo, nullptr, Gb, nullptr, nullptr, nullptr, nullptr,
         65536, 128, 128, 128, 128, 128, 128, 1,
         1, 1, 0, 0, 0, 0, 0, 0, 1.f, M_SIG_F32);
    // tn = LN(t)
    ln_kernel<<<dim3(65536), dim3(128), 128 * 4, stream>>>(Tb, lnow, lnob, BN, 128, 0);
    // z += gate * (tn @ w_o)
    gemm(BN, wo, nullptr, z_out, nullptr, nullptr, Gb, nullptr,
         65536, 128, 128, 128, 128, 128, 128, 1,
         1, 1, 0, 0, 0, 0, 0, 0, 1.f, M_GATE_ADD);
  };
  tri_mul(P[2], P[3], tmo_wp, tmo_wgp, P[6], P[7], tmo_wo, tmo_wgo, 1);
  tri_mul(P[10], P[11], tmi_wp, tmi_wgp, P[14], P[15], tmi_wo, tmi_wgo, 0);

  // ---- triangle attention (start then end) --------------------------------
  auto tri_att = [&](const float* lnw, const float* lnb, bf16* wb, bf16* wq,
                     bf16* wk, bf16* wv, bf16* wg, bf16* wo, int ending) {
    ln_kernel<<<dim3(65536), dim3(128), 128 * 4, stream>>>(z_out, lnw, lnb, XN,
                                                           128, ending ? 256 : 0);
    bf16* q = WD;
    bf16* k = WD + NN * 128;
    bf16* v = WD + 2 * NN * 128;
    gemm(XN, wq, nullptr, nullptr, q, nullptr, nullptr, nullptr,
         65536, 128, 128, 128, 128, 128, 128, 1, 1, 1, 0,0,0,0,0,0, 1.f, M_BF16);
    gemm(XN, wk, nullptr, nullptr, k, nullptr, nullptr, nullptr,
         65536, 128, 128, 128, 128, 128, 128, 1, 1, 1, 0,0,0,0,0,0, 1.f, M_BF16);
    gemm(XN, wv, nullptr, nullptr, v, nullptr, nullptr, nullptr,
         65536, 128, 128, 128, 128, 128, 128, 1, 1, 1, 0,0,0,0,0,0, 1.f, M_BF16);
    gemm(XN, wg, nullptr, Gb, nullptr, nullptr, nullptr, nullptr,
         65536, 128, 128, 128, 128, 128, 128, 1, 1, 1, 0,0,0,0,0,0, 1.f, M_SIG_F32);
    // pair bias: bias16[(q*256+k)*16 + h] (h<4 valid, rest zero)
    gemm(XN, wb, nullptr, B16, nullptr, nullptr, nullptr, nullptr,
         65536, 16, 16, 128, 128, 128, 16, 1, 1, 1, 0,0,0,0,0,0, 1.f, M_F32);
    vt_tri<<<dim3(8388608 / 256), dim3(256), 0, stream>>>(v, VT);

    const float scl = 1.0f / sqrtf(32.f);
    for (int i0 = 0; i0 < 256; i0 += 16) {       // chunk of 16 row-blocks i
      // logits[li][h][q][k] = scl*Q.K^T + bias16
      gemm(q + (long long)i0 * 256 * 128, k + (long long)i0 * 256 * 128, nullptr,
           LOG, nullptr, nullptr, nullptr, B16,
           256, 256, 256, 32, 128, 128, 256, 1,
           64, 4, 256 * 128, 32, 256 * 128, 32, 4LL * 65536, 65536,
           scl, M_LOGITS);
      softmax_kernel<<<dim3(16 * 4 * 256), dim3(256), 0, stream>>>(LOG, PRB);
      // o[q,d] = P.V : out -> Tb[(i*256+q)*128 + h*32 + d]
      gemm(PRB, VT + (long long)i0 * 4 * 32 * 256, nullptr,
           Tb + (long long)i0 * 256 * 128, nullptr, nullptr, nullptr, nullptr,
           256, 32, 32, 256, 256, 256, 128, 1,
           64, 4, 4LL * 65536, 65536, 4LL * 8192, 8192, 256 * 128, 32,
           1.f, M_F32);
    }
    // og = o * gate ; z += og @ w_o (transposed add for 'ending')
    gate_mul<<<dim3(8388608 / 256), dim3(256), 0, stream>>>(Tb, Gb, BN, 8388608);
    gemm(BN, wo, nullptr, z_out, nullptr, nullptr, nullptr, nullptr,
         65536, 128, 128, 128, 128, 128, 128, 1,
         1, 1, 0, 0, 0, 0, 0, 0, 1.f, ending ? M_ADD_T : M_ADD);
  };
  tri_att(P[18], P[19], tas_wb, tas_wq, tas_wk, tas_wv, tas_wg, tas_wo, 0);
  tri_att(P[26], P[27], tae_wb, tae_wq, tae_wk, tae_wv, tae_wg, tae_wo, 1);

  // ---- transitions --------------------------------------------------------
  auto trans = [&](float* xbuf, int R, int D, const float* lnw, const float* lnb,
                   bf16* w1, bf16* w2, bf16* w3) {
    bf16* xn = (D == 128) ? XN : (bf16*)(ws + OFF_MISC + 1 * MB);
    ln_kernel<<<dim3((unsigned)R), dim3((unsigned)D), D * 4, stream>>>(
        xbuf, lnw, lnb, xn, D, 0);
    int DH = 4 * D;
    gemm(xn, w1, w2, nullptr, WD, nullptr, nullptr, nullptr,
         R, DH, DH, D, D, D, DH, 1, 1, 1, 0,0,0,0,0,0, 1.f, M_DUAL_SILU_BF16);
    gemm(WD, w3, nullptr, xbuf, nullptr, nullptr, nullptr, nullptr,
         R, D, D, DH, DH, DH, D, 1, 1, 1, 0,0,0,0,0,0, 1.f, M_ADD);
  };
  trans(z_out, 65536, 128, P[34], P[35], tz_w1, tz_w2, tz_w3);

  // ---- attention with pair bias on s --------------------------------------
  {
    bf16* sn   = (bf16*)(ws + OFF_MISC + 1 * MB);   // [256,384]
    bf16* qpad = (bf16*)(ws + OFF_MISC);            // [256,512]
    bf16* kpad = qpad + 256 * 512;
    ln_kernel<<<dim3(256), dim3(384), 384 * 4, stream>>>(s_out, P[39], P[40], sn, 384, 0);
    zero_bf16<<<dim3((2 * 256 * 512) / 256), dim3(256), 0, stream>>>(qpad, 2 * 256 * 512);
    gemm(sn, at_wq, nullptr, nullptr, qpad, P[42], nullptr, nullptr,
         256, 384, 384, 384, 384, 384, 0, 0, 1, 1, 0,0,0,0,0,0, 1.f, M_PAD_BF16);
    gemm(sn, at_wk, nullptr, nullptr, kpad, nullptr, nullptr, nullptr,
         256, 384, 384, 384, 384, 384, 0, 0, 1, 1, 0,0,0,0,0,0, 1.f, M_PAD_BF16);
    bf16* v = WD;                                   // [256,384] bf16
    gemm(sn, at_wv, nullptr, nullptr, v, nullptr, nullptr, nullptr,
         256, 384, 384, 384, 384, 384, 384, 1, 1, 1, 0,0,0,0,0,0, 1.f, M_BF16);
    gemm(sn, at_wg, nullptr, Gb, nullptr, nullptr, nullptr, nullptr,
         256, 384, 384, 384, 384, 384, 384, 1, 1, 1, 0,0,0,0,0,0, 1.f, M_SIG_F32);
    vt_apb<<<dim3(131072 / 256), dim3(256), 0, stream>>>(v, VT);
    // zb = LN(z) @ wz -> B16 [65536,16]
    ln_kernel<<<dim3(65536), dim3(128), 128 * 4, stream>>>(z_out, P[46], P[47], XN, 128, 0);
    gemm(XN, at_wz, nullptr, B16, nullptr, nullptr, nullptr, nullptr,
         65536, 16, 16, 128, 128, 128, 16, 1, 1, 1, 0,0,0,0,0,0, 1.f, M_F32);
    // logits[h][q][k] = scl*Q.K^T + zb[q,k,h]
    const float scl = 1.0f / sqrtf(24.f);
    gemm(qpad, kpad, nullptr, LOG, nullptr, nullptr, nullptr, B16,
         256, 256, 256, 32, 512, 512, 256, 1,
         16, 16, 0, 32, 0, 32, 0, 65536, scl, M_LOGITS);
    softmax_kernel<<<dim3(16 * 256), dim3(256), 0, stream>>>(LOG, PRB);
    // o[q, h*24+d] = P.V
    gemm(PRB, VT, nullptr, Tb, nullptr, nullptr, nullptr, nullptr,
         256, 24, 32, 256, 256, 256, 384, 1,
         16, 16, 0, 65536, 0, 8192, 0, 24, 1.f, M_F32);
    bf16* og = BN;
    gate_mul<<<dim3((98304 + 255) / 256), dim3(256), 0, stream>>>(Tb, Gb, og, 98304);
    gemm(og, at_wo, nullptr, s_out, nullptr, nullptr, nullptr, nullptr,
         256, 384, 384, 384, 384, 384, 384, 1, 1, 1, 0,0,0,0,0,0, 1.f, M_ADD);
  }

  trans(s_out, 256, 384, P[50], P[51], ts_w1, ts_w2, ts_w3);
}
